// IntraModule1_80169859547838
// MI455X (gfx1250) — matrix operand order repaired and checked by reading, not compiled, not run on hardware
//
#include <hip/hip_runtime.h>
#include <hip/hip_bf16.h>

#define BB 4
#define CCH 64
#define C3 192
#define PP 4096
#define HH 64
#define WW 64

typedef __attribute__((ext_vector_type(16))) __bf16 v16bf;
typedef __attribute__((ext_vector_type(8)))  float  v8f;

#define WMMA_BF16(A,Bf,Acc) __builtin_amdgcn_wmma_f32_16x16x32_bf16(false,(A),false,(Bf),(short)0,(Acc),false,false)

__device__ __forceinline__ float sigmoidf(float v){ return 1.f/(1.f+__expf(-v)); }

// issue an async global->LDS 16-byte copy (ASYNCcnt-tracked, no VGPR data)
__device__ __forceinline__ void async_g2l_b128(unsigned lds_byte_off, const void* gaddr) {
    asm volatile("global_load_async_to_lds_b128 %0, %1, off"
                 :: "v"(lds_byte_off), "v"((unsigned long long)gaddr) : "memory");
}

// ---------------------------------------------------------------- converts
__global__ void k_f32_to_bf16(const float* __restrict__ src, __bf16* __restrict__ dst, int n) {
    int i = blockIdx.x*256 + threadIdx.x;
    if (i < n) dst[i] = (__bf16)src[i];
}

// src: per batch (Cc x P) row-major at src + b*srcBatchStride
// dst: (nBatch, P, Cc) bf16, optional per (b,c) scale
__global__ void k_transpose_scale_bf16(const float* __restrict__ src, __bf16* __restrict__ dst,
                                       const float* __restrict__ scale,
                                       int Cc, long srcBatchStride, int nBatch) {
    long total = (long)nBatch * Cc * PP;
    long i = (long)blockIdx.x*256 + threadIdx.x;
    if (i >= total) return;
    int p = (int)(i & (PP-1));
    long t = i >> 12;
    int c = (int)(t % Cc);
    int b = (int)(t / Cc);
    float v = src[(long)b*srcBatchStride + (long)c*PP + p];
    if (scale) v *= scale[b*Cc + c];
    dst[((long)b*PP + p)*Cc + c] = (__bf16)v;
}

// src f32 (B,C,P) -> dst1 = bf16(v), dst2 = bf16(v*v) same layout
__global__ void k_convert_sq(const float* __restrict__ src, __bf16* __restrict__ d1,
                             __bf16* __restrict__ d2, long n) {
    long i = (long)blockIdx.x*256 + threadIdx.x;
    if (i >= n) return;
    float v = src[i];
    d1[i] = (__bf16)v;
    d2[i] = (__bf16)(v*v);
}

// ---------------------------------------------------------------- WMMA GEMM
// Y[b,Cout,P] = Wt[Cout,Cin] * Xt[b,p,Cin]^T + bias
// one wave per 16(M) x 64(N) tile: A fragment reused across 4 WMMAs
__global__ void k_gemm_wmma(const __bf16* __restrict__ Wt,
                            const __bf16* __restrict__ Xt,
                            const float*  __restrict__ bias,
                            float* __restrict__ Y,
                            int Cout, int Cin) {
    int lane = threadIdx.x & 31;
    int wid  = (int)((blockIdx.x * blockDim.x + threadIdx.x) >> 5);
    int mt = Cout >> 4, pt = PP >> 6;
    if (wid >= BB * mt * pt) return;
    int b   = wid / (mt*pt);
    int rem = wid % (mt*pt);
    int m0 = (rem / pt) * 16;
    int p0 = (rem % pt) * 64;
    int lh = lane & 15, hi = lane >> 4;

    v8f a0 = {}, a1 = {}, a2 = {}, a3 = {};
    const __bf16* wr = Wt + (long)(m0 + lh)*Cin + hi*16;
    const __bf16* xb = Xt + ((long)b*PP + p0 + lh)*Cin + hi*16;
    const long rowst = 16L*Cin;
    for (int kb = 0; kb < Cin; kb += 32) {
        v16bf a;
        #pragma unroll
        for (int e = 0; e < 8; ++e) { a[e] = wr[kb+e]; a[8+e] = wr[kb+8+e]; }
        v16bf b0 = *(const v16bf*)(xb + kb);
        v16bf b1 = *(const v16bf*)(xb + rowst   + kb);
        v16bf b2 = *(const v16bf*)(xb + 2*rowst + kb);
        v16bf b3 = *(const v16bf*)(xb + 3*rowst + kb);
        if (kb + 32 < Cin) __builtin_prefetch(xb + kb + 32, 0, 1);
        a0 = WMMA_BF16(a, b0, a0);
        a1 = WMMA_BF16(a, b1, a1);
        a2 = WMMA_BF16(a, b2, a2);
        a3 = WMMA_BF16(a, b3, a3);
    }
    float* yb = Y + (long)b*Cout*PP;
    #pragma unroll
    for (int r = 0; r < 8; ++r) {
        int m = m0 + hi*8 + r;
        float bv = bias[m];
        float* yr = yb + (long)m*PP + p0 + lh;
        yr[0]  = a0[r] + bv;
        yr[16] = a1[r] + bv;
        yr[32] = a2[r] + bv;
        yr[48] = a3[r] + bv;
    }
}

// ---------------------------------------------------------------- reductions / MLPs
__global__ void k_pool_mean(const float* __restrict__ src, float* __restrict__ out) {
    int bc = blockIdx.x;
    const float* s = src + (long)bc*PP;
    float sum = 0.f;
    for (int i = threadIdx.x; i < PP; i += 256) sum += s[i];
    __shared__ float red[256];
    red[threadIdx.x] = sum; __syncthreads();
    for (int st = 128; st > 0; st >>= 1) {
        if (threadIdx.x < st) red[threadIdx.x] += red[threadIdx.x + st];
        __syncthreads();
    }
    if (threadIdx.x == 0) out[bc] = red[0] * (1.0f/PP);
}

__global__ void k_ca_mlp(const float* __restrict__ avg,
                         const float* __restrict__ w1, const float* __restrict__ b1,
                         const float* __restrict__ w2, const float* __restrict__ b2,
                         float* __restrict__ s, int Cc, int Rh) {
    int b = blockIdx.x, t = threadIdx.x;
    __shared__ float av[192];
    __shared__ float hid[16];
    if (t < Cc) av[t] = avg[b*Cc + t];
    __syncthreads();
    if (t < Rh) {
        float a = b1[t];
        for (int c = 0; c < Cc; ++c) a += w1[t*Cc + c] * av[c];
        hid[t] = fmaxf(a, 0.f);
    }
    __syncthreads();
    if (t < Cc) {
        float a = b2[t];
        for (int h = 0; h < Rh; ++h) a += w2[t*Rh + h] * hid[h];
        s[b*Cc + t] = sigmoidf(a);
    }
}

// ---------------------------------------------------------------- flash attention
// Q,K bf16 (B,P,C);  V,V2 bf16 (B,C,P);  meanT,stdT f32 (B,P,C)
// 8 waves/block share one batch and one key sweep: K/V/V2 32-key tiles are
// staged in LDS via async global->LDS loads, B fragments read from LDS.
__global__ void __launch_bounds__(256) k_attention(
        const __bf16* __restrict__ Q, const __bf16* __restrict__ Kk,
        const __bf16* __restrict__ V, const __bf16* __restrict__ V2,
        float* __restrict__ meanT, float* __restrict__ stdT) {
    int tid   = threadIdx.x;
    int lane  = tid & 31;
    int wslot = tid >> 5;
    int wid   = (blockIdx.x << 3) + wslot;
    int b  = wid >> 8;                    // 256 q-tiles per batch; block-uniform
    int q0 = (wid & 255) * 16;
    int lh = lane & 15, hi = lane >> 4;

    // LDS: K tile 32x64, V tile 64x32, V2 tile 64x32, 8 per-wave p tiles 16x32
    __shared__ __align__(16) __bf16 smem[2048 + 2048 + 2048 + 8*512];
    __bf16* K_s  = smem;
    __bf16* V_s  = smem + 2048;
    __bf16* V2_s = smem + 4096;
    __bf16* pl   = smem + 6144 + wslot*512;
    const unsigned sbase = (unsigned)(size_t)(&smem[0]);

    // A fragments of Q (K-dim = channels 0..31 and 32..63)
    const __bf16* qrow = Q + ((long)b*PP + q0 + lh)*CCH;
    v16bf aq0, aq1;
    #pragma unroll
    for (int e = 0; e < 8; ++e) {
        aq0[e]   = qrow[hi*16 + e];
        aq0[8+e] = qrow[hi*16 + 8 + e];
        aq1[e]   = qrow[32 + hi*16 + e];
        aq1[8+e] = qrow[32 + hi*16 + 8 + e];
    }

    v8f om0={},om1={},om2={},om3={};
    v8f os0={},os1={},os2={},os3={};
    float run_m[8], run_l[8];
    #pragma unroll
    for (int r=0;r<8;++r){ run_m[r] = -1e30f; run_l[r] = 0.f; }

    const __bf16* kbase  = Kk + (long)b*PP*CCH;
    const __bf16* vbase  = V  + (long)b*CCH*PP;
    const __bf16* v2base = V2 + (long)b*CCH*PP;

    // staging indices (each thread copies 3x16B per step)
    const int krow = tid >> 3, kpart = tid & 7;           // K: 32 rows x 64 ch
    const int vrow = tid >> 2, vpart = tid & 3;           // V: 64 ch x 32 keys
    const unsigned ldsK  = sbase + (unsigned)(krow*64 + kpart*8)*2;
    const unsigned ldsV  = sbase + (unsigned)(2048 + vrow*32 + vpart*8)*2;
    const unsigned ldsV2 = sbase + (unsigned)(4096 + vrow*32 + vpart*8)*2;

    for (int kt = 0; kt < PP; kt += 32) {
        // ---- async stage K/V/V2 tiles into LDS
        async_g2l_b128(ldsK,  kbase  + (long)(kt + krow)*CCH + kpart*8);
        async_g2l_b128(ldsV,  vbase  + (long)vrow*PP + kt + vpart*8);
        async_g2l_b128(ldsV2, v2base + (long)vrow*PP + kt + vpart*8);
        asm volatile("s_wait_asynccnt 0x0" ::: "memory");
        __syncthreads();

        // ---- logits (16q x 32k), K-dim = 64 channels, B frags from LDS
        v8f l0 = {}, l1 = {};
        {
            const __bf16* kr0 = K_s + lh*64 + hi*16;
            const __bf16* kr1 = K_s + (16 + lh)*64 + hi*16;
            v16bf bk;
            bk = *(const v16bf*)(kr0);      l0 = WMMA_BF16(aq0, bk, l0);
            bk = *(const v16bf*)(kr0 + 32); l0 = WMMA_BF16(aq1, bk, l0);
            bk = *(const v16bf*)(kr1);      l1 = WMMA_BF16(aq0, bk, l1);
            bk = *(const v16bf*)(kr1 + 32); l1 = WMMA_BF16(aq1, bk, l1);
        }
        // ---- online softmax row stats (row = r + hi*8, cols across 16 lanes)
        float p0v[8], p1v[8];
        #pragma unroll
        for (int r = 0; r < 8; ++r) {
            float m = fmaxf(l0[r], l1[r]);
            m = fmaxf(m, __shfl_xor(m, 1, 32));
            m = fmaxf(m, __shfl_xor(m, 2, 32));
            m = fmaxf(m, __shfl_xor(m, 4, 32));
            m = fmaxf(m, __shfl_xor(m, 8, 32));
            float nm   = fmaxf(run_m[r], m);
            float corr = __expf(run_m[r] - nm);
            run_m[r] = nm;
            p0v[r] = __expf(l0[r] - nm);
            p1v[r] = __expf(l1[r] - nm);
            float sum = p0v[r] + p1v[r];
            sum += __shfl_xor(sum, 1, 32);
            sum += __shfl_xor(sum, 2, 32);
            sum += __shfl_xor(sum, 4, 32);
            sum += __shfl_xor(sum, 8, 32);
            run_l[r] = run_l[r]*corr + sum;
            om0[r]*=corr; om1[r]*=corr; om2[r]*=corr; om3[r]*=corr;
            os0[r]*=corr; os1[r]*=corr; os2[r]*=corr; os3[r]*=corr;
        }
        // ---- reshape p (C/D layout) -> A layout via wave-private LDS tile
        #pragma unroll
        for (int r = 0; r < 8; ++r) {
            int row = r + hi*8;
            pl[row*32 + lh]      = (__bf16)p0v[r];
            pl[row*32 + 16 + lh] = (__bf16)p1v[r];
        }
        asm volatile("s_wait_dscnt 0" ::: "memory");
        v16bf ap;
        #pragma unroll
        for (int e = 0; e < 8; ++e) {
            ap[e]   = pl[lh*32 + hi*16 + e];
            ap[8+e] = pl[lh*32 + hi*16 + 8 + e];
        }
        // ---- PV: fused mean / second-moment accumulation over 4 channel tiles
        int kh = hi*16;
        {
            const __bf16* vr  = V_s  + lh*32 + kh;
            const __bf16* wr2 = V2_s + lh*32 + kh;
            v16bf bv;
            bv = *(const v16bf*)(vr);            om0 = WMMA_BF16(ap, bv, om0);
            bv = *(const v16bf*)(vr + 16*32);    om1 = WMMA_BF16(ap, bv, om1);
            bv = *(const v16bf*)(vr + 32*32);    om2 = WMMA_BF16(ap, bv, om2);
            bv = *(const v16bf*)(vr + 48*32);    om3 = WMMA_BF16(ap, bv, om3);
            bv = *(const v16bf*)(wr2);           os0 = WMMA_BF16(ap, bv, os0);
            bv = *(const v16bf*)(wr2 + 16*32);   os1 = WMMA_BF16(ap, bv, os1);
            bv = *(const v16bf*)(wr2 + 32*32);   os2 = WMMA_BF16(ap, bv, os2);
            bv = *(const v16bf*)(wr2 + 48*32);   os3 = WMMA_BF16(ap, bv, os3);
        }
        __syncthreads();   // protect tiles before next stage overwrites
    }
    // ---- normalize, std = sqrt(relu(E[x^2]-E[x]^2)), store (B,P,C)
    #pragma unroll
    for (int r = 0; r < 8; ++r) {
        int q = q0 + r + hi*8;
        float inv = 1.0f / run_l[r];
        float m0v = om0[r]*inv, m1v = om1[r]*inv, m2v = om2[r]*inv, m3v = om3[r]*inv;
        float s0v = sqrtf(fmaxf(os0[r]*inv - m0v*m0v, 0.f));
        float s1v = sqrtf(fmaxf(os1[r]*inv - m1v*m1v, 0.f));
        float s2v = sqrtf(fmaxf(os2[r]*inv - m2v*m2v, 0.f));
        float s3v = sqrtf(fmaxf(os3[r]*inv - m3v*m3v, 0.f));
        long base = ((long)b*PP + q)*CCH;
        meanT[base +      lh] = m0v;  stdT[base +      lh] = s0v;
        meanT[base + 16 + lh] = m1v;  stdT[base + 16 + lh] = s1v;
        meanT[base + 32 + lh] = m2v;  stdT[base + 32 + lh] = s2v;
        meanT[base + 48 + lh] = m3v;  stdT[base + 48 + lh] = s3v;
    }
}

// ---------------------------------------------------------------- post kernels
__global__ void k_fg_stats(const float* __restrict__ FG, float* __restrict__ fmean,
                           float* __restrict__ frstd) {
    int bc = blockIdx.x;                 // b*64 + c
    int b = bc >> 6, c = bc & 63;
    const float* s = FG + ((long)b*3 + 1)*CCH*PP + (long)c*PP;
    float sum = 0.f, sq = 0.f;
    for (int i = threadIdx.x; i < PP; i += 256) { float v = s[i]; sum += v; sq += v*v; }
    __shared__ float r1[256], r2[256];
    r1[threadIdx.x] = sum; r2[threadIdx.x] = sq; __syncthreads();
    for (int st = 128; st > 0; st >>= 1) {
        if (threadIdx.x < st) { r1[threadIdx.x] += r1[threadIdx.x+st]; r2[threadIdx.x] += r2[threadIdx.x+st]; }
        __syncthreads();
    }
    if (threadIdx.x == 0) {
        float m = r1[0] * (1.0f/PP);
        float var = r2[0] * (1.0f/PP) - m*m;
        fmean[bc] = m;
        frstd[bc] = rsqrtf(fmaxf(var, 0.f) + 1e-5f);
    }
}

__global__ void k_transfer(const float* __restrict__ FG, const float* __restrict__ BG,
                           const float* __restrict__ meanT, const float* __restrict__ stdT,
                           const float* __restrict__ fmean, const float* __restrict__ frstd,
                           float* __restrict__ x) {
    long i = (long)blockIdx.x*256 + threadIdx.x;      // over B*C*P
    if (i >= (long)BB*CCH*PP) return;
    int p = (int)(i & (PP-1));
    long t = i >> 12;
    int c = (int)(t & 63), b = (int)(t >> 6);
    long src = ((long)b*3 + 1)*CCH*PP + (long)c*PP + p;
    float fg = FG[src], bg = BG[src];
    long tix = ((long)b*PP + p)*CCH + c;
    x[i] = stdT[tix] * (fg - fmean[b*64+c]) * frstd[b*64+c] + meanT[tix] + bg;
}

__global__ void k_chanpool(const float* __restrict__ x, float* __restrict__ cavg,
                           float* __restrict__ cmax) {
    int bc = blockIdx.x;
    const float* s = x + (long)bc*PP;
    float sum = 0.f, mx = -1e30f;
    for (int i = threadIdx.x; i < PP; i += 256) { float v = s[i]; sum += v; mx = fmaxf(mx, v); }
    __shared__ float r1[256], r2[256];
    r1[threadIdx.x] = sum; r2[threadIdx.x] = mx; __syncthreads();
    for (int st = 128; st > 0; st >>= 1) {
        if (threadIdx.x < st) { r1[threadIdx.x] += r1[threadIdx.x+st]; r2[threadIdx.x] = fmaxf(r2[threadIdx.x], r2[threadIdx.x+st]); }
        __syncthreads();
    }
    if (threadIdx.x == 0) { cavg[bc] = r1[0]*(1.0f/PP); cmax[bc] = r2[0]; }
}

__global__ void k_cbam_mlp(const float* __restrict__ cavg, const float* __restrict__ cmax,
                           const float* __restrict__ w1, const float* __restrict__ b1,
                           const float* __restrict__ w2, const float* __restrict__ b2,
                           float* __restrict__ s) {
    int b = blockIdx.x, t = threadIdx.x;
    __shared__ float av[64], mxv[64], ha[4], hm[4];
    av[t] = cavg[b*64+t]; mxv[t] = cmax[b*64+t]; __syncthreads();
    if (t < 4) {
        float a = b1[t], m = b1[t];
        for (int c = 0; c < 64; ++c) { a += w1[t*64+c]*av[c]; m += w1[t*64+c]*mxv[c]; }
        ha[t] = fmaxf(a, 0.f); hm[t] = fmaxf(m, 0.f);
    }
    __syncthreads();
    float a = 2.f*b2[t];
    for (int h = 0; h < 4; ++h) a += w2[t*4+h]*(ha[h]+hm[h]);
    s[b*64+t] = sigmoidf(a);
}

__global__ void k_spool(const float* __restrict__ x, const float* __restrict__ s,
                        float* __restrict__ amap, float* __restrict__ mmap) {
    long i = (long)blockIdx.x*256 + threadIdx.x;      // over B*P
    if (i >= (long)BB*PP) return;
    int b = (int)(i >> 12), p = (int)(i & (PP-1));
    float sum = 0.f, mx = -1e30f;
    for (int c = 0; c < 64; ++c) {
        float v = x[((long)b*64 + c)*PP + p] * s[b*64+c];
        sum += v; mx = fmaxf(mx, v);
    }
    amap[i] = sum * (1.0f/64.0f);
    mmap[i] = mx;
}

__global__ void k_spconv(const float* __restrict__ amap, const float* __restrict__ mmap,
                         const float* __restrict__ spw, const float* __restrict__ spb,
                         float* __restrict__ sig) {
    long i = (long)blockIdx.x*256 + threadIdx.x;      // over B*P
    if (i >= (long)BB*PP) return;
    int b = (int)(i >> 12), p = (int)(i & (PP-1));
    int y = p >> 6, xx = p & 63;
    float acc = spb[0];
    for (int dy = -3; dy <= 3; ++dy) {
        int yy = y + dy;
        if (yy < 0 || yy >= HH) continue;
        for (int dx = -3; dx <= 3; ++dx) {
            int xc = xx + dx;
            if (xc < 0 || xc >= WW) continue;
            long idx = ((long)b << 12) + yy*WW + xc;
            int wi = (dy+3)*7 + (dx+3);
            acc += spw[wi]*amap[idx] + spw[49+wi]*mmap[idx];
        }
    }
    sig[i] = sigmoidf(acc);
}

__global__ void k_final(const float* __restrict__ x, const float* __restrict__ s,
                        const float* __restrict__ sig, float* __restrict__ out) {
    long i = (long)blockIdx.x*256 + threadIdx.x;      // over B*C*P
    if (i >= (long)BB*CCH*PP) return;
    int p = (int)(i & (PP-1));
    long t = i >> 12;
    int c = (int)(t & 63), b = (int)(t >> 6);
    out[i] = x[i] * s[b*64+c] * sig[((long)b << 12) + p];
}

// ---------------------------------------------------------------- launcher
extern "C" void kernel_launch(void* const* d_in, const int* in_sizes, int n_in,
                              void* d_out, int out_size, void* d_ws, size_t ws_size,
                              hipStream_t stream) {
    const float* group = (const float*)d_in[0];
    const float* FG    = (const float*)d_in[1];
    const float* BG    = (const float*)d_in[2];
    const float* wg1   = (const float*)d_in[3];
    const float* bg1   = (const float*)d_in[4];
    const float* cag_w1= (const float*)d_in[5];
    const float* cag_b1= (const float*)d_in[6];
    const float* cag_w2= (const float*)d_in[7];
    const float* cag_b2= (const float*)d_in[8];
    const float* wg2   = (const float*)d_in[9];
    const float* bg2   = (const float*)d_in[10];
    const float* wg3   = (const float*)d_in[11];
    const float* bg3   = (const float*)d_in[12];
    const float* wc1   = (const float*)d_in[13];
    const float* bc1   = (const float*)d_in[14];
    const float* cac_w1= (const float*)d_in[15];
    const float* cac_b1= (const float*)d_in[16];
    const float* cac_w2= (const float*)d_in[17];
    const float* cac_b2= (const float*)d_in[18];
    const float* wc2   = (const float*)d_in[19];
    const float* bc2   = (const float*)d_in[20];
    const float* f_w   = (const float*)d_in[21];
    const float* f_b   = (const float*)d_in[22];
    const float* g_w   = (const float*)d_in[23];
    const float* g_b   = (const float*)d_in[24];
    const float* h_w   = (const float*)d_in[25];
    const float* h_b   = (const float*)d_in[26];
    const float* mlp_w1= (const float*)d_in[27];
    const float* mlp_b1= (const float*)d_in[28];
    const float* mlp_w2= (const float*)d_in[29];
    const float* mlp_b2= (const float*)d_in[30];
    const float* sp_w  = (const float*)d_in[31];
    const float* sp_b  = (const float*)d_in[32];

    char* ws = (char*)d_ws;
    size_t off = 0;
    auto alloc = [&](size_t bytes) -> char* {
        char* r = ws + off;
        off += (bytes + 255) & ~(size_t)255;
        return r;
    };

    // bf16 weights
    __bf16* wg1b = (__bf16*)alloc((size_t)C3*C3*2);
    __bf16* wg2b = (__bf16*)alloc((size_t)CCH*C3*2);
    __bf16* wg3b = (__bf16*)alloc((size_t)CCH*CCH*2);
    __bf16* wc1b = (__bf16*)alloc((size_t)CCH*CCH*2);
    __bf16* wc2b = (__bf16*)alloc((size_t)CCH*CCH*2);
    __bf16* fwb  = (__bf16*)alloc((size_t)CCH*CCH*2);
    __bf16* gwb  = (__bf16*)alloc((size_t)CCH*CCH*2);
    __bf16* hwb  = (__bf16*)alloc((size_t)CCH*CCH*2);
    // activation buffers (lifetime-based reuse)
    __bf16* BF1 = (__bf16*)alloc((size_t)BB*PP*C3*2);   // Xt_grp -> Xt1
    __bf16* BF2 = (__bf16*)alloc((size_t)BB*PP*CCH*2);  // Xt2 -> Xt_fg -> Xt_bg
    __bf16* BF3 = (__bf16*)alloc((size_t)BB*PP*CCH*2);  // Xt_c -> Xtc1 -> Xt_fc -> Gk
    __bf16* Fq  = (__bf16*)alloc((size_t)BB*PP*CCH*2);
    __bf16* Hvb = (__bf16*)alloc((size_t)BB*CCH*PP*2);
    __bf16* Hv2b= (__bf16*)alloc((size_t)BB*CCH*PP*2);
    float* FA   = (float*)alloc((size_t)BB*C3*PP*4);    // Y1/Y3/Yc1/Yc2/Yf/Yg/Yh
    float* FB   = (float*)alloc((size_t)BB*CCH*PP*4);   // Y2 -> meanT
    float* FC   = (float*)alloc((size_t)BB*CCH*PP*4);   // stdT
    float* FD   = (float*)alloc((size_t)BB*CCH*PP*4);   // x
    float* g_avg = (float*)alloc(BB*C3*4);
    float* s_g   = (float*)alloc(BB*C3*4);
    float* c_avg = (float*)alloc(BB*CCH*4);
    float* s_c   = (float*)alloc(BB*CCH*4);
    float* fmean = (float*)alloc(BB*CCH*4);
    float* frstd = (float*)alloc(BB*CCH*4);
    float* cavg  = (float*)alloc(BB*CCH*4);
    float* cmax  = (float*)alloc(BB*CCH*4);
    float* s_cb  = (float*)alloc(BB*CCH*4);
    float* amap  = (float*)alloc((size_t)BB*PP*4);
    float* mmap  = (float*)alloc((size_t)BB*PP*4);
    float* sig   = (float*)alloc((size_t)BB*PP*4);
    (void)ws_size; (void)in_sizes; (void)n_in; (void)out_size;

    // weight conversion
    k_f32_to_bf16<<<(C3*C3+255)/256, 256, 0, stream>>>(wg1, wg1b, C3*C3);
    k_f32_to_bf16<<<(CCH*C3+255)/256, 256, 0, stream>>>(wg2, wg2b, CCH*C3);
    k_f32_to_bf16<<<(CCH*CCH+255)/256, 256, 0, stream>>>(wg3, wg3b, CCH*CCH);
    k_f32_to_bf16<<<(CCH*CCH+255)/256, 256, 0, stream>>>(wc1, wc1b, CCH*CCH);
    k_f32_to_bf16<<<(CCH*CCH+255)/256, 256, 0, stream>>>(wc2, wc2b, CCH*CCH);
    k_f32_to_bf16<<<(CCH*CCH+255)/256, 256, 0, stream>>>(f_w, fwb, CCH*CCH);
    k_f32_to_bf16<<<(CCH*CCH+255)/256, 256, 0, stream>>>(g_w, gwb, CCH*CCH);
    k_f32_to_bf16<<<(CCH*CCH+255)/256, 256, 0, stream>>>(h_w, hwb, CCH*CCH);

    const int T192 = BB*C3*PP/256, T64 = BB*CCH*PP/256;
    const int G192 = BB*(C3/16)*(PP/64)/8;   // gemm blocks Cout=192 (16x64 tiles)
    const int G64  = BB*(CCH/16)*(PP/64)/8;  // gemm blocks Cout=64

    // ---- group path: conv1(192x192) -> CA -> conv2(64x192) -> conv3(64x64)
    k_transpose_scale_bf16<<<T192, 256, 0, stream>>>(group, BF1, nullptr, C3, (long)C3*PP, BB);
    k_gemm_wmma<<<G192, 256, 0, stream>>>(wg1b, BF1, bg1, FA, C3, C3);
    k_pool_mean<<<BB*C3, 256, 0, stream>>>(FA, g_avg);
    k_ca_mlp<<<BB, C3, 0, stream>>>(g_avg, cag_w1, cag_b1, cag_w2, cag_b2, s_g, C3, 12);
    k_transpose_scale_bf16<<<T192, 256, 0, stream>>>(FA, BF1, s_g, C3, (long)C3*PP, BB);
    k_gemm_wmma<<<G64, 256, 0, stream>>>(wg2b, BF1, bg2, FB, CCH, C3);
    k_transpose_scale_bf16<<<T64, 256, 0, stream>>>(FB, BF2, nullptr, CCH, (long)CCH*PP, BB);
    k_gemm_wmma<<<G64, 256, 0, stream>>>(wg3b, BF2, bg3, FA, CCH, CCH);
    k_transpose_scale_bf16<<<T64, 256, 0, stream>>>(FA, BF2, nullptr, CCH, (long)CCH*PP, BB); // Xt_fg

    // ---- center path: conv1 -> CA -> conv2
    k_transpose_scale_bf16<<<T64, 256, 0, stream>>>(group + (size_t)CCH*PP, BF3, nullptr, CCH, (long)3*CCH*PP, BB);
    k_gemm_wmma<<<G64, 256, 0, stream>>>(wc1b, BF3, bc1, FA, CCH, CCH);
    k_pool_mean<<<BB*CCH, 256, 0, stream>>>(FA, c_avg);
    k_ca_mlp<<<BB, CCH, 0, stream>>>(c_avg, cac_w1, cac_b1, cac_w2, cac_b2, s_c, CCH, 4);
    k_transpose_scale_bf16<<<T64, 256, 0, stream>>>(FA, BF3, s_c, CCH, (long)CCH*PP, BB);
    k_gemm_wmma<<<G64, 256, 0, stream>>>(wc2b, BF3, bc2, FA, CCH, CCH);
    k_transpose_scale_bf16<<<T64, 256, 0, stream>>>(FA, BF3, nullptr, CCH, (long)CCH*PP, BB); // Xt_fc

    // ---- f/g/h projections
    k_gemm_wmma<<<G64, 256, 0, stream>>>(fwb, BF2, f_b, FA, CCH, CCH);               // Yf from fea_group
    k_transpose_scale_bf16<<<T64, 256, 0, stream>>>(FA, Fq, nullptr, CCH, (long)CCH*PP, BB);
    k_gemm_wmma<<<G64, 256, 0, stream>>>(gwb, BF3, g_b, FA, CCH, CCH);               // Yg from fea_center
    k_transpose_scale_bf16<<<T64, 256, 0, stream>>>(FA, BF3, nullptr, CCH, (long)CCH*PP, BB); // Gk
    k_transpose_scale_bf16<<<T64, 256, 0, stream>>>(BG + (size_t)CCH*PP, BF2, nullptr, CCH, (long)3*CCH*PP, BB); // Xt_bg
    k_gemm_wmma<<<G64, 256, 0, stream>>>(hwb, BF2, h_b, FA, CCH, CCH);               // Yh
    k_convert_sq<<<T64, 256, 0, stream>>>(FA, Hvb, Hv2b, (long)BB*CCH*PP);

    // ---- flash attention: meanT=FB, stdT=FC (B,P,C)
    k_attention<<<BB*(PP/16)/8, 256, 0, stream>>>(Fq, BF3, Hvb, Hv2b, FB, FC);

    // ---- transfer + CBAM
    k_fg_stats<<<BB*CCH, 256, 0, stream>>>(FG, fmean, frstd);
    k_transfer<<<T64, 256, 0, stream>>>(FG, BG, FB, FC, fmean, frstd, FD);
    k_chanpool<<<BB*CCH, 256, 0, stream>>>(FD, cavg, cmax);
    k_cbam_mlp<<<BB, CCH, 0, stream>>>(cavg, cmax, mlp_w1, mlp_b1, mlp_w2, mlp_b2, s_cb);
    k_spool<<<(BB*PP+255)/256, 256, 0, stream>>>(FD, s_cb, amap, mmap);
    k_spconv<<<(BB*PP+255)/256, 256, 0, stream>>>(amap, mmap, sp_w, sp_b, sig);
    k_final<<<T64, 256, 0, stream>>>(FD, s_cb, sig, (float*)d_out);
}
